// MultiHeadAttention_41455024341166
// MI455X (gfx1250) — compile-verified, matrix operand order checked by reading, not run on hardware
//
#include <hip/hip_runtime.h>
#include <hip/hip_bf16.h>

// ---------------------------------------------------------------------------
// CDNA5 (gfx1250, wave32) fused MHA block:
//   1) qkv_rope_gemm : X[8192,2048] @ Wqkv[2048,6144] -> bf16 q,k (RoPE'd), v
//      (double-buffered LDS, v_wmma_f32_16x16x32_bf16, fused RoPE epilogue)
//   2) flash_attn    : causal online-softmax attention; K tiles prefetched by
//      the Tensor Data Mover (tensor_load_to_lds, TENSORcnt-overlapped),
//      bf16 WMMA, f32 accumulate
//   3) out_proj_gemm : ctx[8192,2048](bf16) @ Wo[2048,2048] -> out f32
// ---------------------------------------------------------------------------

typedef __attribute__((ext_vector_type(16))) __bf16 v16bf;
typedef __attribute__((ext_vector_type(8)))  float  v8f;
typedef __attribute__((ext_vector_type(4)))  unsigned int v4u;
typedef __attribute__((ext_vector_type(8)))  int v8i;
typedef __attribute__((ext_vector_type(4)))  int v4i;

#if __has_builtin(__builtin_amdgcn_tensor_load_to_lds) && \
    __has_builtin(__builtin_amdgcn_s_wait_tensorcnt)
#define HAVE_TDM 1
#else
#define HAVE_TDM 0
#endif

union FragB16 { v16bf v; uint4 q[2]; unsigned short h[16]; };

static __device__ __forceinline__ unsigned short f2bf(float f) {
  unsigned int u = __float_as_uint(f);
  u += 0x7FFFu + ((u >> 16) & 1u);          // round-to-nearest-even
  return (unsigned short)(u >> 16);
}
static __device__ __forceinline__ unsigned int f2bf_pk(float lo, float hi) {
  return (unsigned int)f2bf(lo) | ((unsigned int)f2bf(hi) << 16);
}

// A/B 16x32 bf16 fragment: per lane two contiguous 8-element (16B) runs at
// +kb and +16+kb, kb = 8*(lane>=16).  Works for LDS and global sources.
static __device__ __forceinline__ v16bf frag_ld(const unsigned short* p, int hi) {
  FragB16 f;
  f.q[0] = *reinterpret_cast<const uint4*>(p + hi * 8);
  f.q[1] = *reinterpret_cast<const uint4*>(p + 16 + hi * 8);
  return f.v;
}

// ---------------------------------------------------------------------------
// Cooperative tile loaders (256-thread blocks) with f32 -> bf16 conversion.
// ---------------------------------------------------------------------------
static __device__ __forceinline__ void load_x_tile(
    const float* __restrict__ X, int rb, int k0, int tid,
    unsigned short (*__restrict__ As)[40]) {
  #pragma unroll
  for (int i = 0; i < 4; ++i) {             // 128x32 f32 -> bf16 (packed b32)
    int f = tid + i * 256, r = f >> 3, c4 = (f & 7) * 4;
    float4 xv = *reinterpret_cast<const float4*>(X + (size_t)(rb + r) * 2048 + k0 + c4);
    uint2 p;
    p.x = f2bf_pk(xv.x, xv.y);
    p.y = f2bf_pk(xv.z, xv.w);
    *reinterpret_cast<uint2*>(&As[r][c4]) = p;
  }
}

template <int N>
static __device__ __forceinline__ void load_w_tile(
    const float* __restrict__ W, int k0, int nb, int tid,
    unsigned short (*__restrict__ Bt)[40]) {
  #pragma unroll
  for (int i = 0; i < 4; ++i) {             // 32xN-slice f32 -> bf16 transposed
    int f = tid + i * 256, kr = f >> 5, c4 = (f & 31) * 4;
    float4 wv = *reinterpret_cast<const float4*>(W + (size_t)(k0 + kr) * N + nb + c4);
    Bt[c4 + 0][kr] = f2bf(wv.x); Bt[c4 + 1][kr] = f2bf(wv.y);
    Bt[c4 + 2][kr] = f2bf(wv.z); Bt[c4 + 3][kr] = f2bf(wv.w);
  }
}

// ---------------------------------------------------------------------------
// Kernel 1: QKV projection GEMM with fused RoPE epilogue.
// grid (6144/128, 8192/128), block 256 (8 waves; each wave 32x64 of the tile)
// ---------------------------------------------------------------------------
__global__ __launch_bounds__(256) void qkv_rope_gemm(
    const float* __restrict__ X, const float* __restrict__ W,
    unsigned short* __restrict__ qws, unsigned short* __restrict__ kws,
    unsigned short* __restrict__ vws)
{
  constexpr int M = 2048, N = 6144, S = 2048, D = 128, H = 16;
  const int nb = blockIdx.x * 128;
  const int rb = blockIdx.y * 128;
  const int tid   = threadIdx.x;
  const int wave  = tid >> 5, lane = tid & 31;
  const int lan16 = lane & 15, hi = lane >> 4;
  const int wr = wave & 3, wc = wave >> 2;

  __shared__ unsigned short As[2][128][40];   // double buffered, 16B-aligned rows
  __shared__ unsigned short Bt[2][128][40];   // B transposed: [n][k]

  v8f acc[2][4];
  #pragma unroll
  for (int i = 0; i < 2; ++i)
    #pragma unroll
    for (int j = 0; j < 4; ++j) acc[i][j] = {};

  load_x_tile(X, rb, 0, tid, As[0]);
  load_w_tile<N>(W, 0, nb, tid, Bt[0]);
  __syncthreads();

  for (int kt = 0; kt < M / 32; ++kt) {
    const int buf = kt & 1;
    if (kt + 1 < M / 32) {                    // prefetch next tile into other buf
      load_x_tile(X, rb, (kt + 1) * 32, tid, As[buf ^ 1]);
      load_w_tile<N>(W, (kt + 1) * 32, nb, tid, Bt[buf ^ 1]);
    }
    v16bf a[2], b[4];
    #pragma unroll
    for (int i = 0; i < 2; ++i) a[i] = frag_ld(&As[buf][wr * 32 + i * 16 + lan16][0], hi);
    #pragma unroll
    for (int j = 0; j < 4; ++j) b[j] = frag_ld(&Bt[buf][wc * 64 + j * 16 + lan16][0], hi);
    #pragma unroll
    for (int i = 0; i < 2; ++i)
      #pragma unroll
      for (int j = 0; j < 4; ++j)
        acc[i][j] = __builtin_amdgcn_wmma_f32_16x16x32_bf16(
            false, a[i], false, b[j], (short)0, acc[i][j], false, false);
    __syncthreads();
  }

  // Epilogue: RoPE on q,k (t<2).  C layout: col = lane&15, row = v + 8*(lane>=16).
  // RoPE partner (d^1) lives in lane^1 -> one shfl_xor.
  const float kLn1e4o64 = 0.14391156831212784f;   // ln(10000)/64
  #pragma unroll
  for (int i = 0; i < 2; ++i) {
    #pragma unroll
    for (int j = 0; j < 4; ++j) {
      const int n = nb + wc * 64 + j * 16 + lan16;
      const int t = n >> 11, rem = n & 2047, h = rem >> 7, d = rem & 127;
      unsigned short* dst = (t == 0) ? qws : (t == 1) ? kws : vws;
      const float freq = __expf(-(float)(d >> 1) * kLn1e4o64);
      const float sgn  = (d & 1) ? 1.0f : -1.0f;
      #pragma unroll
      for (int v = 0; v < 8; ++v) {
        const int r  = rb + wr * 32 + i * 16 + v + hi * 8;
        const int s  = r & (S - 1);
        const int bb = r >> 11;
        float c = acc[i][j][v];
        float partner = __shfl_xor(c, 1, 32);   // computed uniformly
        float o;
        if (t < 2) {
          float ang = (float)s * freq;
          o = c * __cosf(ang) + partner * __sinf(ang) * sgn;
        } else {
          o = c;
        }
        dst[(((size_t)(bb * H + h) * S) + s) * D + d] = f2bf(o);
      }
    }
  }
}

// ---------------------------------------------------------------------------
// Flash-attention tile loaders.
// ---------------------------------------------------------------------------
static __device__ __forceinline__ void load_k_tile(
    const unsigned short* __restrict__ kp, int ks0, int tid,
    unsigned short (*__restrict__ Ks)[136]) {
  #pragma unroll
  for (int i = 0; i < 8; ++i) {             // 64x128 bf16, b128 copies
    int f = tid + i * 128, r = f >> 4, c8 = (f & 15) * 8;
    *reinterpret_cast<uint4*>(&Ks[r][c8]) =
        *reinterpret_cast<const uint4*>(kp + (size_t)(ks0 + r) * 128 + c8);
  }
}
static __device__ __forceinline__ void load_v_tile(
    const unsigned short* __restrict__ vp, int ks0, int tid,
    unsigned short (*__restrict__ Vt)[72]) {
  #pragma unroll
  for (int i = 0; i < 8; ++i) {             // transpose V into LDS [d][key]
    int f = tid + i * 128, r = f >> 4, c8 = (f & 15) * 8;
    uint4 v4 = *reinterpret_cast<const uint4*>(vp + (size_t)(ks0 + r) * 128 + c8);
    const unsigned short* e = reinterpret_cast<const unsigned short*>(&v4);
    #pragma unroll
    for (int j = 0; j < 8; ++j) Vt[c8 + j][r] = e[j];
  }
}

#if HAVE_TDM
// Tensor Data Mover: async-copy one 64x128 bf16 K tile (row stride 128 elems)
// into an LDS tile with a 136-element row stride, expressed via TDM padding:
// 64 DWORDs (128 bf16) per row, then +4 DWORDs (8 bf16) pad.
// This toolchain's builtin takes 6 args:
//   (uint32x4 g0, int32x8 g1, int32x4 g2, int32x4 g3, int32x8 extra, i32 cpol)
static __device__ __forceinline__ void tdm_load_k_tile(
    const unsigned short* gsrc, unsigned short* ldst) {
  const unsigned long long ga = (unsigned long long)(size_t)gsrc;   // byte addr
  const unsigned lds = (unsigned)(size_t)ldst;                      // LDS offset
  v4u g0 = {
    1u,                                            // count=1 valid descriptor
    lds,                                           // lds_addr  (bits 63:32)
    (unsigned)ga,                                  // global_addr[31:0]
    ((unsigned)(ga >> 32) & 0x01FFFFFFu) | 0x80000000u  // addr[56:32] | type=2
  };
  v8i g1;
  g1[0] = (1 << 16)      // data_size = 2 bytes
        | (1 << 20)      // pad_enable
        | (5 << 22)      // pad_interval code 5 = 64 DWORDs before each pad
        | (3 << 25);     // pad_amount  code 3 = 4 DWORDs of pad
  g1[1] = (int)(128u << 16);   // tensor_dim0 = 128   (bits 79:48, low half)
  g1[2] = (int)(64u << 16);    // tensor_dim1 = 64    (bits 111:80, low half)
  g1[3] = (int)(128u << 16);   // tile_dim0  = 128    (bits 127:112)
  g1[4] = 64;                  // tile_dim1 = 64, tile_dim2 = 0
  g1[5] = 128;                 // tensor_dim0_stride = 128 elements
  g1[6] = 0;
  g1[7] = 0;
  v4i zero4 = {0, 0, 0, 0};            // 2-D tensor: groups 2/3 unused
  v8i zero8 = {0, 0, 0, 0, 0, 0, 0, 0};
  __builtin_amdgcn_tensor_load_to_lds(g0, g1, zero4, zero4, zero8, 0);
}
#endif

// ---------------------------------------------------------------------------
// Kernel 2: causal flash attention.
// grid (S/64, B*H), block 128 (4 waves; each wave owns 16 query rows).
// K tiles double-buffered + TDM-prefetched; V transposed manually (TDM cannot
// transpose); P relayout via per-wave LDS staging.  LDS = 60KB.
// ---------------------------------------------------------------------------
__global__ __launch_bounds__(128) void flash_attn(
    const unsigned short* __restrict__ qws, const unsigned short* __restrict__ kws,
    const unsigned short* __restrict__ vws, unsigned short* __restrict__ ctxws)
{
  constexpr int S = 2048, D = 128, HD = 2048;
  const int qt = blockIdx.x;
  const int bh = blockIdx.y;
  const int b = bh >> 4, h = bh & 15;
  const int tid = threadIdx.x, wave = tid >> 5, lane = tid & 31;
  const int lan16 = lane & 15, hi = lane >> 4;

  __shared__ unsigned short Ks[2][64][136];  // K tiles, double buffered
  __shared__ unsigned short Vt[128][72];     // V tile transposed [d][key]
  __shared__ unsigned short Ps[4][16][64];   // per-wave P staging (C->A relayout)

  const size_t plane = (size_t)bh * S * D;
  const unsigned short* qb = qws + plane;
  const unsigned short* kp = kws + plane;
  const unsigned short* vp = vws + plane;

  const int qrow0 = qt * 64 + wave * 16;

  // Q fragments for this wave's 16 rows (4 chunks of K=32 across D=128)
  v16bf Aq[4];
  {
    const unsigned short* qr = qb + (size_t)(qrow0 + lan16) * D;
    #pragma unroll
    for (int c = 0; c < 4; ++c) Aq[c] = frag_ld(qr + c * 32, hi);
  }

  v8f ctx[8];
  #pragma unroll
  for (int dt = 0; dt < 8; ++dt) ctx[dt] = {};
  float m_run[8], l_run[8];
  #pragma unroll
  for (int v = 0; v < 8; ++v) { m_run[v] = -1e30f; l_run[v] = 0.0f; }

  const int nkt = qt + 1;                    // causal: key tiles up to diagonal

  // Kick off K tile 0.
#if HAVE_TDM
  if (wave == 0) tdm_load_k_tile(kp, &Ks[0][0][0]);
#else
  load_k_tile(kp, 0, tid, Ks[0]);
#endif

  for (int kt = 0; kt < nkt; ++kt) {
    const int buf = kt & 1;
    const int ks0 = kt * 64;

    load_v_tile(vp, ks0, tid, Vt);           // prev readers done (loop-end barrier)
    if (kt + 1 < nkt) {                      // prefetch next K tile
#if HAVE_TDM
      if (wave == 0) tdm_load_k_tile(kp + (size_t)(ks0 + 64) * D, &Ks[buf ^ 1][0][0]);
#else
      load_k_tile(kp, ks0 + 64, tid, Ks[buf ^ 1]);
#endif
    }
#if HAVE_TDM
    // Oldest TDM (this tile's K) must be done; newest may stay in flight.
    if (wave == 0) {
      if (kt + 1 < nkt) __builtin_amdgcn_s_wait_tensorcnt(1);
      else              __builtin_amdgcn_s_wait_tensorcnt(0);
    }
#endif
    __syncthreads();

    if (ks0 <= qrow0 + 15) {                 // wave has unmasked keys here
      // ---- S = Q @ K^T : 4 col-tiles x 4 K-chunks = 16 WMMA ----
      v8f sc[4];
      #pragma unroll
      for (int ct = 0; ct < 4; ++ct) {
        v8f a = {};
        const unsigned short* kcp = &Ks[buf][ct * 16 + lan16][0];
        #pragma unroll
        for (int c = 0; c < 4; ++c) {
          v16bf bf = frag_ld(kcp + c * 32, hi);
          a = __builtin_amdgcn_wmma_f32_16x16x32_bf16(false, Aq[c], false, bf,
                                                      (short)0, a, false, false);
        }
        sc[ct] = a;
      }
      // ---- scale + causal mask ----
      const float scale = 0.08838834764831845f;   // 1/sqrt(128)
      #pragma unroll
      for (int ct = 0; ct < 4; ++ct) {
        const int kcol = ks0 + ct * 16 + lan16;
        #pragma unroll
        for (int v = 0; v < 8; ++v) {
          const int qr = qrow0 + v + hi * 8;
          float sv = sc[ct][v] * scale;
          sc[ct][v] = (kcol <= qr) ? sv : -1e30f;
        }
      }
      // ---- online softmax (row stats via 16-lane xor reductions) ----
      #pragma unroll
      for (int v = 0; v < 8; ++v) {
        float mv = fmaxf(fmaxf(sc[0][v], sc[1][v]), fmaxf(sc[2][v], sc[3][v]));
        mv = fmaxf(mv, __shfl_xor(mv, 1, 32));
        mv = fmaxf(mv, __shfl_xor(mv, 2, 32));
        mv = fmaxf(mv, __shfl_xor(mv, 4, 32));
        mv = fmaxf(mv, __shfl_xor(mv, 8, 32));
        const float mnew = fmaxf(m_run[v], mv);
        const float corr = __expf(m_run[v] - mnew);
        m_run[v] = mnew;
        l_run[v] *= corr;
        #pragma unroll
        for (int dt = 0; dt < 8; ++dt) ctx[dt][v] *= corr;
      }
      #pragma unroll
      for (int ct = 0; ct < 4; ++ct)
        #pragma unroll
        for (int v = 0; v < 8; ++v)
          sc[ct][v] = __expf(sc[ct][v] - m_run[v]);
      #pragma unroll
      for (int v = 0; v < 8; ++v) {
        float rs = (sc[0][v] + sc[1][v]) + (sc[2][v] + sc[3][v]);
        rs += __shfl_xor(rs, 1, 32);
        rs += __shfl_xor(rs, 2, 32);
        rs += __shfl_xor(rs, 4, 32);
        rs += __shfl_xor(rs, 8, 32);
        l_run[v] += rs;
      }
      // ---- C-layout -> A-layout via per-wave LDS staging ----
      #pragma unroll
      for (int ct = 0; ct < 4; ++ct)
        #pragma unroll
        for (int v = 0; v < 8; ++v)
          Ps[wave][v + hi * 8][ct * 16 + lan16] = f2bf(sc[ct][v]);
      // ---- ctx += P @ V : 8 d-tiles x 2 K-chunks = 16 WMMA ----
      #pragma unroll
      for (int dt = 0; dt < 8; ++dt) {
        const unsigned short* vc = &Vt[dt * 16 + lan16][0];
        #pragma unroll
        for (int kc = 0; kc < 2; ++kc) {
          v16bf fa = frag_ld(&Ps[wave][lan16][kc * 32], hi);
          v16bf fb = frag_ld(vc + kc * 32, hi);
          ctx[dt] = __builtin_amdgcn_wmma_f32_16x16x32_bf16(
              false, fa, false, fb, (short)0, ctx[dt], false, false);
        }
      }
    }
    __syncthreads();                         // Vt free for next iteration
  }

  // ---- epilogue: normalize, store bf16 ctx as [B*S, H*D] ----
  float inv[8];
  #pragma unroll
  for (int v = 0; v < 8; ++v) inv[v] = 1.0f / l_run[v];
  #pragma unroll
  for (int dt = 0; dt < 8; ++dt) {
    #pragma unroll
    for (int v = 0; v < 8; ++v) {
      const int s = qrow0 + v + hi * 8;
      const size_t row = (size_t)b * S + s;
      const int col = h * D + dt * 16 + lan16;
      ctxws[row * HD + col] = f2bf(ctx[dt][v] * inv[v]);
    }
  }
}

// ---------------------------------------------------------------------------
// Kernel 3: output projection GEMM. ctx(bf16)[8192,2048] @ Wo[2048,2048] -> f32
// grid (2048/128, 8192/128), block 256, double-buffered LDS.
// ---------------------------------------------------------------------------
static __device__ __forceinline__ void load_ctx_tile(
    const unsigned short* __restrict__ Cx, int rb, int k0, int tid,
    unsigned short (*__restrict__ As)[40]) {
  #pragma unroll
  for (int i = 0; i < 2; ++i) {              // already bf16: b128 copies
    int f = tid + i * 256, r = f >> 2, c8 = (f & 3) * 8;
    *reinterpret_cast<uint4*>(&As[r][c8]) =
        *reinterpret_cast<const uint4*>(Cx + (size_t)(rb + r) * 2048 + k0 + c8);
  }
}

__global__ __launch_bounds__(256) void out_proj_gemm(
    const unsigned short* __restrict__ Cx, const float* __restrict__ Wo,
    float* __restrict__ out)
{
  constexpr int K = 2048, N = 2048;
  const int nb = blockIdx.x * 128;
  const int rb = blockIdx.y * 128;
  const int tid   = threadIdx.x;
  const int wave  = tid >> 5, lane = tid & 31;
  const int lan16 = lane & 15, hi = lane >> 4;
  const int wr = wave & 3, wc = wave >> 2;

  __shared__ unsigned short As[2][128][40];
  __shared__ unsigned short Bt[2][128][40];

  v8f acc[2][4];
  #pragma unroll
  for (int i = 0; i < 2; ++i)
    #pragma unroll
    for (int j = 0; j < 4; ++j) acc[i][j] = {};

  load_ctx_tile(Cx, rb, 0, tid, As[0]);
  load_w_tile<N>(Wo, 0, nb, tid, Bt[0]);
  __syncthreads();

  for (int kt = 0; kt < K / 32; ++kt) {
    const int buf = kt & 1;
    if (kt + 1 < K / 32) {
      load_ctx_tile(Cx, rb, (kt + 1) * 32, tid, As[buf ^ 1]);
      load_w_tile<N>(Wo, (kt + 1) * 32, nb, tid, Bt[buf ^ 1]);
    }
    v16bf a[2], b[4];
    #pragma unroll
    for (int i = 0; i < 2; ++i) a[i] = frag_ld(&As[buf][wr * 32 + i * 16 + lan16][0], hi);
    #pragma unroll
    for (int j = 0; j < 4; ++j) b[j] = frag_ld(&Bt[buf][wc * 64 + j * 16 + lan16][0], hi);
    #pragma unroll
    for (int i = 0; i < 2; ++i)
      #pragma unroll
      for (int j = 0; j < 4; ++j)
        acc[i][j] = __builtin_amdgcn_wmma_f32_16x16x32_bf16(
            false, a[i], false, b[j], (short)0, acc[i][j], false, false);
    __syncthreads();
  }

  #pragma unroll
  for (int i = 0; i < 2; ++i)
    #pragma unroll
    for (int j = 0; j < 4; ++j) {
      const int n = nb + wc * 64 + j * 16 + lan16;
      #pragma unroll
      for (int v = 0; v < 8; ++v) {
        const int r = rb + wr * 32 + i * 16 + v + hi * 8;
        out[(size_t)r * N + n] = acc[i][j][v];
      }
    }
}

// ---------------------------------------------------------------------------
extern "C" void kernel_launch(void* const* d_in, const int* in_sizes, int n_in,
                              void* d_out, int out_size, void* d_ws, size_t ws_size,
                              hipStream_t stream) {
  (void)in_sizes; (void)n_in; (void)out_size; (void)ws_size;
  constexpr int B = 4, S = 2048, H = 16, D = 128;
  constexpr size_t PLANE = (size_t)B * H * S * D;      // 16,777,216 bf16 elems

  const float* x    = (const float*)d_in[0];
  const float* Wqkv = (const float*)d_in[1];
  const float* Wo   = (const float*)d_in[2];
  float* out = (float*)d_out;

  unsigned short* qws = (unsigned short*)d_ws;         // [B][H][S][D] bf16
  unsigned short* kws = qws + PLANE;
  unsigned short* vws = kws + PLANE;
  unsigned short* cws = vws + PLANE;                   // [B*S][H*D]  bf16

  qkv_rope_gemm<<<dim3(6144 / 128, (B * S) / 128), 256, 0, stream>>>(
      x, Wqkv, qws, kws, vws);
  flash_attn<<<dim3(S / 64, B * H), 128, 0, stream>>>(qws, kws, vws, cws);
  out_proj_gemm<<<dim3(2048 / 128, (B * S) / 128), 256, 0, stream>>>(
      cws, Wo, out);
}
